// ReflectionModel_64914135712079
// MI455X (gfx1250) — matrix operand order repaired and checked by reading, not compiled, not run on hardware
//
#include <hip/hip_runtime.h>

// ---------------------------------------------------------------------------
// Fused MARL attention-critic forward for MI455X (gfx1250, wave32, WMMA+TDM).
// B=8192, A=8, OBS=128, NACT=32, H=256, D=64.  One WG = 2 batch groups.
//   prep_weights_kernel: f32 weights -> f16 WMMA-B fragments in d_ws (once).
//   fused kernel: TDM bulk-loads the 77824B fragment blob into LDS while the
//   waves stage activations; 3 GEMMs + softmax + weighted sum, all in LDS.
// ---------------------------------------------------------------------------

typedef __attribute__((ext_vector_type(16))) _Float16 v16h;
typedef __attribute__((ext_vector_type(8)))  _Float16 v8h;
typedef __attribute__((ext_vector_type(8)))  float    v8f;
typedef __attribute__((ext_vector_type(4)))  unsigned int tdm_v4u;
typedef __attribute__((ext_vector_type(8)))  int          tdm_v8i;
typedef __attribute__((ext_vector_type(4)))  int          tdm_v4i;

#define NBATCH 8192
#define NAG    8
#define OBSD   128
#define NACT   32
#define HDIM   256
#define DATT   64

// ---- weight fragment blob layout (bytes, same in d_ws and in LDS) ---------
#define WB_WQ   0              // 8*4*32*16 f16 = 32768
#define WB_WK   32768          // 1*4*32*16 f16 = 4096
#define WB_WV1  36864          // 9*4*32*16 f16 = 36864
#define WB_WV2  73728          // 2*2*32*16 f16 = 4096
#define W_BYTES 77824          // = 19456 dwords

// ---- LDS layout (bytes) ---------------------------------------------------
#define OFF_SQ    0                    // 16*64  f32 = 4096
#define OFF_SK    4096                 // 128*64 f32 = 32768
#define OFF_SATTN 36864                // 16*8   f32 = 512
#define OFF_W     37376                // weight blob (TDM target), 77824
#define OFF_S     (OFF_W + W_BYTES)    // 115200: activation scratch
#define OFF_AQ    (OFF_S)              // 16*256 f16 = 8192   (phase 1)
#define OFF_AK    (OFF_S + 8192)       // 128*32 f16 = 8192   (phase 1)
#define OFF_AV    (OFF_S)              // 128*288 f16 = 73728 (phase 3)
#define OFF_HID   (OFF_S + 73728)      // 128*64 f16 = 16384  (phase 3)
#define LDS_BYTES (OFF_S + 73728 + 16384)   // 205312

#if defined(__gfx1250__) && __has_builtin(__builtin_amdgcn_tensor_load_to_lds) \
                         && __has_builtin(__builtin_amdgcn_s_wait_tensorcnt)
#define USE_TDM 1
#else
#define USE_TDM 0
#endif

// WMMA f16 wrapper: D = A*B + C, f32 accumulate.
__device__ __forceinline__ v8f wmma_f16(v16h a, v16h b, v8f c) {
    return __builtin_amdgcn_wmma_f32_16x16x32_f16(false, a, false, b,
                                                  (short)0, c, false, false);
}

// A fragment (16x32 f16 tile) from LDS, row-major with stride `lda` halfs.
// ISA layout: lanes 0-15 row M=lane hold K {kb+0..7, kb+16..23};
//             lanes 16-31 row M=lane-16 hold K {kb+8..15, kb+24..31}.
__device__ __forceinline__ v16h load_a_frag(const _Float16* lds, int rowBase,
                                            int lda, int kb, int lane) {
    int m  = lane & 15;
    int hi = (lane >> 4) & 1;
    const _Float16* p = lds + (rowBase + m) * lda + kb + hi * 8;
    v8h lo = *(const v8h*)p;
    v8h hv = *(const v8h*)(p + 16);
    v16h r;
#pragma unroll
    for (int t = 0; t < 8; ++t) { r[t] = lo[t]; r[t + 8] = hv[t]; }
    return r;
}

// B fragment, pre-swizzled: 16 contiguous halfs per lane per fragment.
__device__ __forceinline__ v16h load_b_frag(const _Float16* base, int fragIdx,
                                            int lane) {
    const _Float16* p = base + ((size_t)fragIdx * 32 + lane) * 16;
    v8h lo = *(const v8h*)p;
    v8h hv = *(const v8h*)(p + 8);
    v16h r;
#pragma unroll
    for (int t = 0; t < 8; ++t) { r[t] = lo[t]; r[t + 8] = hv[t]; }
    return r;
}

// Pre-swizzle W[nOut][ktot] (f32 row-major) into per-lane WMMA-B fragments:
// frag (kc,nt), lane L: col n = nt*16 + (L&15), K = kc*32 + (L<16?0:8) + (t<8?t:t+8).
__device__ __forceinline__ void fill_w_frags_g(_Float16* dst, const float* w,
                                               int ktot, int nkc, int nnt,
                                               int t0, int nth) {
    int nslots = nkc * nnt * 32;
    for (int s = t0; s < nslots; s += nth) {
        int lane = s & 31;
        int frag = s >> 5;
        int nt   = frag % nnt;
        int kc   = frag / nnt;
        int n    = nt * 16 + (lane & 15);
        int koff = kc * 32 + ((lane & 16) ? 8 : 0);
        const float* src = w + (size_t)n * ktot + koff;
        _Float16*    d   = dst + (size_t)s * 16;
#pragma unroll
        for (int t = 0; t < 16; ++t) d[t] = (_Float16)src[t < 8 ? t : t + 8];
    }
}

__global__ __launch_bounds__(256) void prep_weights_kernel(
    const float* __restrict__ wq_w, const float* __restrict__ wk_w,
    const float* __restrict__ wv1_w, const float* __restrict__ wv2_w,
    _Float16* __restrict__ ws) {
    int t0  = blockIdx.x * blockDim.x + threadIdx.x;
    int nth = gridDim.x * blockDim.x;
    fill_w_frags_g(ws + WB_WQ / 2,  wq_w,  2 * OBSD,    8, 4, t0, nth);
    fill_w_frags_g(ws + WB_WK / 2,  wk_w,  NACT,        1, 4, t0, nth);
    fill_w_frags_g(ws + WB_WV1 / 2, wv1_w, HDIM + NACT, 9, 4, t0, nth);
    fill_w_frags_g(ws + WB_WV2 / 2, wv2_w, DATT,        2, 2, t0, nth);
}

__global__ __launch_bounds__(256) void fused_marl_attn_kernel(
    const float* __restrict__ o,      const float* __restrict__ onext,
    const float* __restrict__ hbuf,   const float* __restrict__ actions,
    const _Float16* __restrict__ wfrag,
    const float* __restrict__ wq_b,   const float* __restrict__ wk_b,
    const float* __restrict__ wv1_b,  const float* __restrict__ wv2_b,
    float* __restrict__ out) {
    extern __shared__ char smem[];
    float*    sQ    = (float*)(smem + OFF_SQ);     // [16][64]
    float*    sK    = (float*)(smem + OFF_SK);     // [128][64]
    float*    sAttn = (float*)(smem + OFF_SATTN);  // [16][8]
    _Float16* sWq   = (_Float16*)(smem + OFF_W + WB_WQ);
    _Float16* sWk   = (_Float16*)(smem + OFF_W + WB_WK);
    _Float16* sWv1  = (_Float16*)(smem + OFF_W + WB_WV1);
    _Float16* sWv2  = (_Float16*)(smem + OFF_W + WB_WV2);
    _Float16* sAq   = (_Float16*)(smem + OFF_AQ);  // [16][256]
    _Float16* sAk   = (_Float16*)(smem + OFF_AK);  // [128][32]
    _Float16* sAv   = (_Float16*)(smem + OFF_AV);  // [128][288]
    _Float16* sHid  = (_Float16*)(smem + OFF_HID); // [128][64]

    const int tid  = threadIdx.x;
    const int wav  = tid >> 5;
    const int lane = tid & 31;
    const int hi   = (lane >> 4) & 1;
    const int wg   = blockIdx.x;
    const int n0   = wg * 16;    // first (b,i) row
    const int g0   = wg * 128;   // first (b,i,j) value row

    // ------- Kick off TDM: stream the 77824B weight-fragment blob to LDS ---
#if USE_TDM
    if (wav == 0) {
        const unsigned ldsW = (unsigned)(size_t)(smem + OFF_W);
        const unsigned long long ga = (unsigned long long)(size_t)wfrag;
        const unsigned N = W_BYTES / 4;            // 19456 dwords
        tdm_v4u g0d;
        g0d[0] = 1u;                                        // count=1
        g0d[1] = ldsW;                                      // lds_addr
        g0d[2] = (unsigned)(ga & 0xFFFFFFFFu);              // global_addr lo
        g0d[3] = (unsigned)((ga >> 32) & 0x1FFFFFFu) | (2u << 30); // hi|type=2
        tdm_v8i g1d;
        g1d[0] = (int)(2u << 16);                   // data_size = 4B
        g1d[1] = (int)((N & 0xFFFFu) << 16);        // tensor_dim0[15:0]
        g1d[2] = (int)((N >> 16) | (1u << 16));     // tensor_dim0 hi | tdim1=1
        g1d[3] = (int)((N & 0xFFFFu) << 16);        // tile_dim0 = N
        g1d[4] = 1;                                 // tile_dim1=1, tile_dim2=0
        g1d[5] = (int)N;                            // tensor_dim0_stride lo
        g1d[6] = 0;
        g1d[7] = 0;
        tdm_v4i gz4 = {};
        tdm_v8i gz8 = {};
        __builtin_amdgcn_tensor_load_to_lds(g0d, g1d, gz4, gz4, gz8, 0);
    }
#endif

    // ---------------- Phase 1 fill: query/key activations ------------------
    for (int f = tid; f < 1024; f += 256) {          // sAq: 16x256 f16
        int row = f >> 6, col = (f & 63) * 4;
        int ng  = n0 + row;
        const float* src = (col < OBSD)
            ? (o     + (size_t)ng * OBSD + col)
            : (onext + (size_t)ng * OBSD + (col - OBSD));
        float4 v = *(const float4*)src;
        _Float16* d = sAq + row * 256 + col;
        d[0] = (_Float16)v.x; d[1] = (_Float16)v.y;
        d[2] = (_Float16)v.z; d[3] = (_Float16)v.w;
    }
    for (int f = tid; f < 1024; f += 256) {          // sAk: 128x32 f16
        int row = f >> 3, col = (f & 7) * 4;
        float4 v = *(const float4*)(actions + (size_t)(g0 + row) * NACT + col);
        _Float16* d = sAk + row * NACT + col;
        d[0] = (_Float16)v.x; d[1] = (_Float16)v.y;
        d[2] = (_Float16)v.z; d[3] = (_Float16)v.w;
    }
#if USE_TDM
    if (wav == 0) __builtin_amdgcn_s_wait_tensorcnt(0);
#else
    {   // fallback: bulk-copy fragment blob ws -> LDS (no conversion needed)
        const uint4* src = (const uint4*)wfrag;
        uint4* dst = (uint4*)(smem + OFF_W);
        for (int i = tid; i < W_BYTES / 16; i += 256) dst[i] = src[i];
    }
#endif
    __syncthreads();

    // ---------------- Phase 1 compute: query (waves 0-3), key (4-7) --------
    if (wav < 4) {
        const int nt = wav;
        v8f acc = {};
#pragma unroll
        for (int kc = 0; kc < 8; ++kc) {
            v16h a = load_a_frag(sAq, 0, 256, kc * 32, lane);
            acc = wmma_f16(a, load_b_frag(sWq, kc * 4 + nt, lane), acc);
        }
        int ncol = nt * 16 + (lane & 15);
        float bias = wq_b[ncol];
#pragma unroll
        for (int r = 0; r < 8; ++r)
            sQ[(r + 8 * hi) * DATT + ncol] = acc[r] + bias;
    } else {
#pragma unroll
        for (int sub = 0; sub < 2; ++sub) {
            int mt = (wav - 4) * 2 + sub;
            v16h a = load_a_frag(sAk, mt * 16, NACT, 0, lane);
#pragma unroll
            for (int nt = 0; nt < 4; ++nt) {
                v8f acc = {};
                acc = wmma_f16(a, load_b_frag(sWk, nt, lane), acc);
                int ncol = nt * 16 + (lane & 15);
                float bias = wk_b[ncol];
#pragma unroll
                for (int r = 0; r < 8; ++r)
                    sK[(mt * 16 + r + 8 * hi) * DATT + ncol] = acc[r] + bias;
            }
        }
    }
    __syncthreads();

    // ---------------- Phase 2: masked scores + softmax ---------------------
    if (tid < 128) {
        int nl = tid >> 3, j = tid & 7;
        const float* qp = sQ + nl * DATT;
        const float* kp = sK + (nl * 8 + j) * DATT;
        float s = 0.f;
#pragma unroll
        for (int d = 0; d < DATT; ++d) s += qp[d] * kp[d];
        s *= 0.125f;                       // 1/sqrt(64)
        if (j == (nl & 7)) s = -1e9f;      // self-mask (i == j)
        sAttn[nl * 8 + j] = s;
    }
    __syncthreads();
    if (tid < 16) {
        float* p = sAttn + tid * 8;
        float mx = p[0];
#pragma unroll
        for (int j = 1; j < 8; ++j) mx = fmaxf(mx, p[j]);
        float e[8], sum = 0.f;
#pragma unroll
        for (int j = 0; j < 8; ++j) { e[j] = __expf(p[j] - mx); sum += e[j]; }
        float inv = 1.f / sum;
#pragma unroll
        for (int j = 0; j < 8; ++j) p[j] = e[j] * inv;
    }
    __syncthreads();

    // ---------------- Phase 3 fill: value MLP inputs -----------------------
    for (int f = tid; f < 8192; f += 256) {          // h part: 128x256
        int row = f >> 6, col = (f & 63) * 4;
        int bl = row >> 6, j = row & 7;
        int hrow = (wg * 2 + bl) * 8 + j;            // h reused 8x via LDS
        float4 v = *(const float4*)(hbuf + (size_t)hrow * HDIM + col);
        _Float16* d = sAv + row * 288 + col;
        d[0] = (_Float16)v.x; d[1] = (_Float16)v.y;
        d[2] = (_Float16)v.z; d[3] = (_Float16)v.w;
    }
    for (int f = tid; f < 1024; f += 256) {          // action part: 128x32
        int row = f >> 3, col = (f & 7) * 4;
        float4 v = *(const float4*)(actions + (size_t)(g0 + row) * NACT + col);
        _Float16* d = sAv + row * 288 + HDIM + col;
        d[0] = (_Float16)v.x; d[1] = (_Float16)v.y;
        d[2] = (_Float16)v.z; d[3] = (_Float16)v.w;
    }
    __syncthreads();

    // ---------------- Phase 3a: hid = relu([h|a] @ wv1^T + b1) -------------
    v8f hc[4] = {v8f{}, v8f{}, v8f{}, v8f{}};
#pragma unroll
    for (int kc = 0; kc < 9; ++kc) {
        v16h a = load_a_frag(sAv, wav * 16, 288, kc * 32, lane);
#pragma unroll
        for (int nt = 0; nt < 4; ++nt)
            hc[nt] = wmma_f16(a, load_b_frag(sWv1, kc * 4 + nt, lane), hc[nt]);
    }
#pragma unroll
    for (int nt = 0; nt < 4; ++nt) {
        int ncol = nt * 16 + (lane & 15);
        float bias = wv1_b[ncol];
#pragma unroll
        for (int r = 0; r < 8; ++r) {
            float v = fmaxf(hc[nt][r] + bias, 0.f);
            sHid[(wav * 16 + r + 8 * hi) * DATT + ncol] = (_Float16)v;
        }
    }
    __syncthreads();

    // ---------------- Phase 3b: value = hid @ wv2^T; attn-weighted sum -----
    v8f q0 = {}, q1 = {};
#pragma unroll
    for (int kc = 0; kc < 2; ++kc) {
        v16h a = load_a_frag(sHid, wav * 16, DATT, kc * 32, lane);
        q0 = wmma_f16(a, load_b_frag(sWv2, kc * 2 + 0, lane), q0);
        q1 = wmma_f16(a, load_b_frag(sWv2, kc * 2 + 1, lane), q1);
    }
    // C-layout: VGPR r holds M = r + 8*hi, i.e. j = r, output pair = hi.
    int nloc = wav * 2 + hi;
    const float* aw = sAttn + nloc * 8;
    float r0 = 0.f, r1 = 0.f;
#pragma unroll
    for (int r = 0; r < 8; ++r) { r0 += q0[r] * aw[r]; r1 += q1[r] * aw[r]; }
    int c0 = lane & 15, c1 = 16 + (lane & 15);
    size_t obase = (size_t)(n0 + nloc) * NACT;
    out[obase + c0] = r0 + wv2_b[c0];
    out[obase + c1] = r1 + wv2_b[c1];
}

extern "C" void kernel_launch(void* const* d_in, const int* in_sizes, int n_in,
                              void* d_out, int out_size, void* d_ws,
                              size_t ws_size, hipStream_t stream) {
    const float* o       = (const float*)d_in[0];
    const float* onext   = (const float*)d_in[1];
    const float* hbuf    = (const float*)d_in[2];
    const float* actions = (const float*)d_in[3];
    const float* wq_w    = (const float*)d_in[4];
    const float* wq_b    = (const float*)d_in[5];
    const float* wk_w    = (const float*)d_in[6];
    const float* wk_b    = (const float*)d_in[7];
    const float* wv1_w   = (const float*)d_in[8];
    const float* wv1_b   = (const float*)d_in[9];
    const float* wv2_w   = (const float*)d_in[10];
    const float* wv2_b   = (const float*)d_in[11];
    float* out = (float*)d_out;
    _Float16* wfrag = (_Float16*)d_ws;   // 77824B fragment blob

    (void)hipFuncSetAttribute((const void*)fused_marl_attn_kernel,
                              hipFuncAttributeMaxDynamicSharedMemorySize,
                              LDS_BYTES);

    prep_weights_kernel<<<dim3(4), dim3(256), 0, stream>>>(
        wq_w, wk_w, wv1_w, wv2_w, wfrag);

    dim3 grid(NBATCH / 2);   // 4096 WGs, each owns 2 batch groups
    dim3 block(256);         // 8 wave32 waves
    fused_marl_attn_kernel<<<grid, block, LDS_BYTES, stream>>>(
        o, onext, hbuf, actions, wfrag, wq_b, wk_b, wv1_b, wv2_b, out);
}
